// M5_19164144074971
// MI455X (gfx1250) — matrix operand order repaired and checked by reading, not compiled, not run on hardware
//
#include <hip/hip_runtime.h>
#include <hip/hip_bf16.h>

typedef __attribute__((ext_vector_type(16))) _Float16 v16h;
typedef __attribute__((ext_vector_type(8)))  float    v8f;

#define HID     128
#define NEG_SL  0.01f
#define BN_EPS  1e-5f

union BFrag {            // 16-bit B/A fragment viewed as 8 pair-words
  v16h     h;
  unsigned u[8];
};
union PairW {
  _Float16 f[2];
  unsigned u;
};

// ---------------------------------------------------------------------------
// fp32 -> f16 conversion into K-pair-packed layout:
//   dst[(k>>1)*cols + c] = { src[k][c] (lo), src[k+1][c] (hi) },  zero K-pad.
// This matches the CDNA5 16-bit B-fragment register layout so B fragments can
// be loaded as 8x b32 instead of 16x u16 + half-packing.
// ---------------------------------------------------------------------------
__global__ __launch_bounds__(256)
void cvt_pack_pair(const float* __restrict__ src, unsigned* __restrict__ dst,
                   int rows, int cols, int dst_rows) {
  int total = (dst_rows >> 1) * cols;
  for (int i = blockIdx.x * blockDim.x + threadIdx.x; i < total;
       i += gridDim.x * blockDim.x) {
    int k2 = i / cols;
    int c  = i - k2 * cols;
    int k0 = 2 * k2;
    PairW p;
    p.f[0] = (k0 < rows)     ? (_Float16)src[k0 * cols + c]       : (_Float16)0.f;
    p.f[1] = (k0 + 1 < rows) ? (_Float16)src[(k0 + 1) * cols + c] : (_Float16)0.f;
    dst[i] = p.u;
  }
}

// ---------------------------------------------------------------------------
// Edge kernel: per 16-edge tile (one wave):
//   emb = edge_attr(16, K-padded to 32) @ ew(f16) + eb   via WMMA
//   msg = relu(x[src] + emb);  atomicAdd into agg[dst]
// GUARD=false: full tile, branch-free scatter (all but the last tile).
// ---------------------------------------------------------------------------
template <bool GUARD>
__device__ __forceinline__
void edge_body(long tile, int lane,
               const float* __restrict__ eattr,
               const int* __restrict__ srcI, const int* __restrict__ dstI,
               const unsigned* __restrict__ ewp,   // packed [16][ic] pair-words
               const float* __restrict__ eb,
               const float* __restrict__ x, int xstride,
               float* agg, int ic, int ne) {
  int rlo  = lane & 15;
  int koff = (lane >> 4) * 16;

  // --- A fragment: 16 edges x K(=32, top half zero) ---
  v16h a;
  {
    long e  = tile + rlo;
    bool ok = !GUARD || (e < ne);
    const float* ap = eattr + (ok ? e : 0) * 16;
#pragma unroll
    for (int i = 0; i < 8; ++i) {
      int kb = koff + 2 * i;
#pragma unroll
      for (int j = 0; j < 2; ++j) {
        int k   = kb + j;
        float v = (ok && k < 16) ? ap[k] : 0.f;
        a[2 * i + j] = (_Float16)v;
      }
    }
  }

  // rows this lane owns in the C/D fragment (for gather/scatter)
  int mo = (lane >> 4) * 8;
  int sA[8], dA[8];
  bool okA[8];
#pragma unroll
  for (int i = 0; i < 8; ++i) {
    long e  = tile + mo + i;
    okA[i] = !GUARD || (e < ne);
    sA[i]  = okA[i] ? srcI[e] : 0;
    dA[i]  = okA[i] ? dstI[e] : 0;
  }

  int kb2h = (lane >> 4) * 8;     // pair-word row base in packed ew
  int nt   = ic >> 4;
  for (int t = 0; t < nt; ++t) {
    int col = t * 16 + rlo;
    BFrag b;
#pragma unroll
    for (int i = 0; i < 8; ++i) b.u[i] = ewp[(kb2h + i) * ic + col];

    v8f c;
    float bv = eb[col];
#pragma unroll
    for (int i = 0; i < 8; ++i) c[i] = bv;
    c = __builtin_amdgcn_wmma_f32_16x16x32_f16(false, a, false, b.h, (short)0,
                                               c, false, false);
    // fused gather + relu + scatter-add
#pragma unroll
    for (int i = 0; i < 8; ++i) {
      if (!GUARD || okA[i]) {
        float v = c[i] + x[(long)sA[i] * xstride + col];
        v = fmaxf(v, 0.f);
        atomicAdd(&agg[(long)dA[i] * HID + col], v);
      }
    }
  }
}

__global__ __launch_bounds__(256)
void edge_msg_kernel(const float* __restrict__ eattr,
                     const int* __restrict__ srcI, const int* __restrict__ dstI,
                     const unsigned* __restrict__ ewp,
                     const float* __restrict__ eb,
                     const float* __restrict__ x, int xstride,
                     float* agg, int ic, int ne) {
  int wave = threadIdx.x >> 5;
  int lane = threadIdx.x & 31;
  long tile = ((long)blockIdx.x * 8 + wave) * 16;
  if (tile >= ne) return;

  // prefetch next block's edge_attr stream (emits global_prefetch_b8)
  __builtin_prefetch(eattr + (tile + 128) * 16 + lane, 0, 1);

  if (tile + 16 <= ne)
    edge_body<false>(tile, lane, eattr, srcI, dstI, ewp, eb, x, xstride, agg, ic, ne);
  else
    edge_body<true>(tile, lane, eattr, srcI, dstI, ewp, eb, x, xstride, agg, ic, ne);
}

// ---------------------------------------------------------------------------
// Node GEMM: Y[N,128] = H @ W + bias.
//   MODE 0: H = (1+eps)*xin + agg   (fp32 in, f16 convert in-register)
//   MODE 1: H = tin                 (f16 BN output)
// W (pair-packed f16) staged into LDS with async global->LDS B128 loads
// (ASYNCcnt path), then consumed as 8x ds_load_b32 per B fragment.
// Each wave computes a 16x128 strip (8 WMMA accumulators), 8 waves/block.
// NOTE: yout may alias agg (each row is read, then written, by its own wave).
// ---------------------------------------------------------------------------
template <int MODE, bool GUARD>
__device__ __forceinline__
void gemm_body(int rowbase, int lane, const unsigned* WlU,
               const float* __restrict__ xin, const float* agg,
               const float* __restrict__ epsp,
               const _Float16* __restrict__ tin,
               const float* __restrict__ bias,
               float* yout, int K, int xstride, int n) {
  int nlo = lane & 15;
  float epsv = 0.f;
  if (MODE == 0) epsv = 1.f + epsp[0];

  v8f c[8];
#pragma unroll
  for (int t = 0; t < 8; ++t) {
    float bv = bias[t * 16 + nlo];
#pragma unroll
    for (int i = 0; i < 8; ++i) c[t][i] = bv;
  }

  int arow = rowbase + nlo;
  bool rok = !GUARD || (arow < n);
  int koff = (lane >> 4) * 16;

  for (int kt = 0; kt < K; kt += 32) {
    // A fragment
    v16h a;
#pragma unroll
    for (int i = 0; i < 8; ++i) {
      int kb = koff + 2 * i;
#pragma unroll
      for (int j = 0; j < 2; ++j) {
        int k = kt + kb + j;
        float v = 0.f;
        if (rok) {
          if (MODE == 0)
            v = epsv * xin[(long)arow * xstride + k] + agg[(long)arow * HID + k];
          else
            v = (float)tin[(long)arow * HID + k];
        }
        a[2 * i + j] = (_Float16)v;
      }
    }
    int kb2h = (kt >> 1) + (lane >> 4) * 8;   // pair-word row base
#pragma unroll
    for (int t = 0; t < 8; ++t) {
      int col = t * 16 + nlo;
      BFrag b;
#pragma unroll
      for (int i = 0; i < 8; ++i) b.u[i] = WlU[(kb2h + i) * HID + col];
      c[t] = __builtin_amdgcn_wmma_f32_16x16x32_f16(false, a, false, b.h,
                                                    (short)0, c[t], false, false);
    }
  }

  int mo2 = (lane >> 4) * 8;
#pragma unroll
  for (int t = 0; t < 8; ++t) {
#pragma unroll
    for (int i = 0; i < 8; ++i) {
      int row = rowbase + mo2 + i;
      if (!GUARD || row < n) yout[(long)row * HID + t * 16 + nlo] = c[t][i];
    }
  }
}

template <int MODE>
__global__ __launch_bounds__(256)
void node_gemm_kernel(const float* __restrict__ xin, const float* agg,
                      const float* __restrict__ epsp,
                      const _Float16* __restrict__ tin,
                      const unsigned* __restrict__ Wp,  // packed (K/2)xHID words
                      const float* __restrict__ bias,
                      float* yout, int K, int xstride, int n) {
  __shared__ unsigned WlU[64 * HID];   // up to 32 KB of the 320 KB/WGP LDS

  // async global->LDS staging of the weight tile (ASYNCcnt tracked)
  {
    int chunks = (K * HID * 2) / 16;                // 16B chunks
    const char* gbase = (const char*)Wp;
    unsigned ldsbase = (unsigned)(uintptr_t)(&WlU[0]);  // LDS aperture: low 32b
    for (int idx = threadIdx.x; idx < chunks; idx += blockDim.x) {
      unsigned lds = ldsbase + idx * 16;
      const char* g = gbase + (size_t)idx * 16;
      asm volatile("global_load_async_to_lds_b128 %0, %1, off"
                   :: "v"(lds), "v"(g)
                   : "memory");
    }
    asm volatile("s_wait_asynccnt 0x0" ::: "memory");
    __syncthreads();
  }

  int wave = threadIdx.x >> 5;
  int lane = threadIdx.x & 31;
  int rowbase = (blockIdx.x * 8 + wave) * 16;
  if (rowbase >= n) return;

  if (rowbase + 16 <= n)
    gemm_body<MODE, false>(rowbase, lane, WlU, xin, agg, epsp, tin, bias, yout,
                           K, xstride, n);
  else
    gemm_body<MODE, true>(rowbase, lane, WlU, xin, agg, epsp, tin, bias, yout,
                          K, xstride, n);
}

// ---------------------------------------------------------------------------
// Per-column sum / sum-of-squares for BatchNorm batch statistics.
// ---------------------------------------------------------------------------
__global__ __launch_bounds__(128)
void stats_kernel(const float* __restrict__ y, float* __restrict__ stats, int n) {
  int c = threadIdx.x;          // 128 threads = 128 columns
  float s = 0.f, q = 0.f;
  for (int r = blockIdx.x; r < n; r += gridDim.x) {
    float v = y[(long)r * HID + c];
    s += v;
    q += v * v;
  }
  atomicAdd(&stats[c], s);
  atomicAdd(&stats[HID + c], q);
}

// ---------------------------------------------------------------------------
// BN (training-mode, biased var) + LeakyReLU; writes f16 and/or fp32 outputs.
// ---------------------------------------------------------------------------
__global__ __launch_bounds__(256)
void bn_act_kernel(const float* __restrict__ y, const float* __restrict__ stats,
                   const float* __restrict__ g, const float* __restrict__ be,
                   _Float16* outh, float* outf, int n) {
  long total = (long)n * HID;
  float invN = 1.f / (float)n;
  for (long i = (long)blockIdx.x * blockDim.x + threadIdx.x; i < total;
       i += (long)gridDim.x * blockDim.x) {
    int c = (int)(i & (HID - 1));
    float mean = stats[c] * invN;
    float var  = stats[HID + c] * invN - mean * mean;
    float v = (y[i] - mean) * rsqrtf(var + BN_EPS) * g[c] + be[c];
    v = v > 0.f ? v : NEG_SL * v;
    if (outh) outh[i] = (_Float16)v;
    if (outf) outf[i] = v;
  }
}

// ---------------------------------------------------------------------------
// global_add_pool via atomics.
// ---------------------------------------------------------------------------
__global__ __launch_bounds__(256)
void pool_kernel(const float* __restrict__ x, const int* __restrict__ batch,
                 float* pool, int n) {
  long total = (long)n * HID;
  for (long i = (long)blockIdx.x * blockDim.x + threadIdx.x; i < total;
       i += (long)gridDim.x * blockDim.x) {
    int node = (int)(i >> 7);
    int c    = (int)(i & (HID - 1));
    atomicAdd(&pool[(long)batch[node] * HID + c], x[i]);
  }
}

// ---------------------------------------------------------------------------
// logits = [x, pool[batch]] @ wf + bf ; softmax over 2 classes.
// ---------------------------------------------------------------------------
__global__ __launch_bounds__(256)
void final_kernel(const float* __restrict__ x, const float* __restrict__ pool,
                  const int* __restrict__ batch, const float* __restrict__ wf,
                  const float* __restrict__ bf, float* __restrict__ out, int n) {
  int node = blockIdx.x * blockDim.x + threadIdx.x;
  if (node >= n) return;
  const float* xr = x + (long)node * HID;
  const float* pr = pool + (long)batch[node] * HID;
  float l0 = bf[0], l1 = bf[1];
#pragma unroll 8
  for (int k = 0; k < HID; ++k) {
    float xv = xr[k];
    l0 += xv * wf[2 * k];
    l1 += xv * wf[2 * k + 1];
  }
#pragma unroll 8
  for (int k = 0; k < HID; ++k) {
    float pv = pr[k];
    l0 += pv * wf[2 * (HID + k)];
    l1 += pv * wf[2 * (HID + k) + 1];
  }
  float m  = fmaxf(l0, l1);
  float e0 = __expf(l0 - m), e1 = __expf(l1 - m);
  float inv = 1.f / (e0 + e1);
  out[2 * node]     = e0 * inv;
  out[2 * node + 1] = e1 * inv;
}

// ---------------------------------------------------------------------------
// Driver
// ---------------------------------------------------------------------------
extern "C" void kernel_launch(void* const* d_in, const int* in_sizes, int n_in,
                              void* d_out, int out_size, void* d_ws, size_t ws_size,
                              hipStream_t stream) {
  const float* x0    = (const float*)d_in[0];
  const int*   ei    = (const int*)d_in[1];
  const float* ea    = (const float*)d_in[2];
  const int*   batch = (const int*)d_in[3];
  const int N = in_sizes[0] / 32;
  const int E = in_sizes[1] / 2;
  const int G = 4096;
  const float* wf = (const float*)d_in[37];
  const float* bf = (const float*)d_in[38];

  // workspace layout (256B aligned): ~130 MB total
  char* base = (char*)d_ws;
  size_t off = 0;
  auto alloc = [&](size_t bytes) -> void* {
    off = (off + 255) & ~(size_t)255;
    void* p = base + off;
    off += bytes;
    return p;
  };
  float*    xbuf  = (float*)alloc((size_t)N * HID * sizeof(float));
  float*    agg   = (float*)alloc((size_t)N * HID * sizeof(float));
  float*    ybuf  = agg;   // deliberate safe alias (see node_gemm_kernel note)
  _Float16* tbuf  = (_Float16*)alloc((size_t)N * HID * sizeof(_Float16));
  float*    pool  = (float*)alloc((size_t)G * HID * sizeof(float));
  float*    stats = (float*)alloc(2 * HID * sizeof(float));
  unsigned* ewp[3];
  unsigned* w1p[3];
  unsigned* w2p[3];
  for (int l = 0; l < 3; ++l) {
    ewp[l] = (unsigned*)alloc(16 * HID * sizeof(unsigned));        // 32/2 x ic
    w1p[l] = (unsigned*)alloc((HID / 2) * HID * sizeof(unsigned));
    w2p[l] = (unsigned*)alloc((HID / 2) * HID * sizeof(unsigned));
  }

  // convert weights to pair-packed f16 (K-pad edge weights 16 -> 32)
  for (int l = 0; l < 3; ++l) {
    int ic = (l == 0) ? 32 : HID;
    const float* ew = (const float*)d_in[4 + 11 * l + 1];
    const float* w1 = (const float*)d_in[4 + 11 * l + 3];
    const float* w2 = (const float*)d_in[4 + 11 * l + 7];
    cvt_pack_pair<<<64, 256, 0, stream>>>(ew, ewp[l], 16, ic, 32);
    cvt_pack_pair<<<64, 256, 0, stream>>>(w1, w1p[l], ic, HID, ic);
    cvt_pack_pair<<<64, 256, 0, stream>>>(w2, w2p[l], HID, HID, HID);
  }

  const float* xcur = x0;
  int xstride = 32;
  int gemmBlocks = (N + 127) / 128;
  int edgeBlocks = (E + 127) / 128;

  for (int l = 0; l < 3; ++l) {
    int ic = (l == 0) ? 32 : HID;
    const float* epsp = (const float*)d_in[4 + 11 * l + 0];
    const float* eb   = (const float*)d_in[4 + 11 * l + 2];
    const float* b1   = (const float*)d_in[4 + 11 * l + 4];
    const float* g1   = (const float*)d_in[4 + 11 * l + 5];
    const float* be1  = (const float*)d_in[4 + 11 * l + 6];
    const float* b2   = (const float*)d_in[4 + 11 * l + 8];
    const float* g2   = (const float*)d_in[4 + 11 * l + 9];
    const float* be2  = (const float*)d_in[4 + 11 * l + 10];

    // message passing: agg = segment_sum(relu(x[src] + edge_attr@ew + eb), dst)
    hipMemsetAsync(agg, 0, (size_t)N * HID * sizeof(float), stream);
    edge_msg_kernel<<<edgeBlocks, 256, 0, stream>>>(ea, ei, ei + E, ewp[l], eb,
                                                    xcur, xstride, agg, ic, E);

    // h = (1+eps)*x + agg ; Y1 = h @ w1 + b1
    hipMemsetAsync(stats, 0, 2 * HID * sizeof(float), stream);
    node_gemm_kernel<0><<<gemmBlocks, 256, 0, stream>>>(
        xcur, agg, epsp, nullptr, w1p[l], b1, ybuf, ic, xstride, N);
    stats_kernel<<<1024, 128, 0, stream>>>(ybuf, stats, N);
    bn_act_kernel<<<4096, 256, 0, stream>>>(ybuf, stats, g1, be1, tbuf, nullptr, N);

    // Y2 = t @ w2 + b2 ; x = lrelu(bn(Y2))
    hipMemsetAsync(stats, 0, 2 * HID * sizeof(float), stream);
    node_gemm_kernel<1><<<gemmBlocks, 256, 0, stream>>>(
        nullptr, nullptr, nullptr, tbuf, w2p[l], b2, ybuf, HID, HID, N);
    stats_kernel<<<1024, 128, 0, stream>>>(ybuf, stats, N);
    bn_act_kernel<<<4096, 256, 0, stream>>>(ybuf, stats, g2, be2, nullptr, xbuf, N);

    xcur = xbuf;
    xstride = HID;
  }

  // global_add_pool + readout + softmax
  hipMemsetAsync(pool, 0, (size_t)G * HID * sizeof(float), stream);
  pool_kernel<<<4096, 256, 0, stream>>>(xbuf, batch, pool, N);
  final_kernel<<<(N + 255) / 256, 256, 0, stream>>>(xbuf, pool, batch, wf, bf,
                                                    (float*)d_out, N);
}
